// LSTM_3_81183471829674
// MI455X (gfx1250) — 8-bit matrix operand order repaired and checked by reading, compiled, not run on hardware
//
#include <hip/hip_runtime.h>

typedef int   v16i __attribute__((ext_vector_type(16)));
typedef float v8f  __attribute__((ext_vector_type(8)));

#define S_LEN 2048
#define H_DIM 256
#define MTILE 16

__device__ __forceinline__ unsigned int fp8_e4m3(float x) {
  unsigned int u = __float_as_uint(x);
  unsigned int s = (u >> 24) & 0x80u;
  float ax = fabsf(x);
  if (!(ax < 448.0f)) return s | 0x7Eu;
  if (ax < 0.0009765625f) return s;
  if (ax < 0.015625f) {
    int q = (int)(ax * 512.0f + 0.5f);
    return s | (unsigned int)q;
  }
  unsigned int au = __float_as_uint(ax);
  unsigned int lsb = (au >> 20) & 1u;
  au += 0x0007FFFFu + lsb;
  int e = (int)((au >> 23) & 0xFFu) - 127;
  unsigned int m = (au >> 20) & 0x7u;
  if (e > 8) return s | 0x7Eu;
  return s | ((unsigned int)(e + 7) << 3) | m;
}

__device__ __forceinline__ unsigned int pk4_fp8(float a, float b, float c, float d) {
  return fp8_e4m3(a) | (fp8_e4m3(b) << 8) | (fp8_e4m3(c) << 16) | (fp8_e4m3(d) << 24);
}

__device__ __forceinline__ float sigm(float x) { return 1.0f / (1.0f + __expf(-x)); }
__device__ __forceinline__ float tanh_fast(float x) {
  float e = __expf(2.0f * x);
  return (e - 1.0f) / (e + 1.0f);
}

__global__ __launch_bounds__(256)
__attribute__((amdgpu_waves_per_eu(2, 2)))
void lstm_persistent_fp8(const float* __restrict__ x,    const int*   __restrict__ mask,
                         const float* __restrict__ feat, const float* __restrict__ stat,
                         const float* __restrict__ wih,  const float* __restrict__ whh,
                         const float* __restrict__ bih,  const float* __restrict__ bhh,
                         const float* __restrict__ w1,   const float* __restrict__ b1,
                         const float* __restrict__ w2,   const float* __restrict__ b2,
                         float* __restrict__ out)
{
  __shared__ unsigned int h8[2][1024];
  __shared__ unsigned int wFp[8][2][2][32][16];
  __shared__ float wih_l[1024][6];
  __shared__ float bias_l[1024];
  __shared__ float w1s_l[256][10];
  __shared__ float b1_l[256];
  __shared__ float w2_l[256];
  __shared__ float fullbuf[16 * 8];
  __shared__ float sbuf[16 * 12];
  __shared__ float y_buf[16];
  __shared__ float y_acc[16];

  const int tid = threadIdx.x;
  const int w   = tid >> 5;
  const int l   = tid & 31;
  const int g   = l >> 4;
  const int ln  = l & 15;
  const int b0  = blockIdx.x * MTILE;

  for (int i = tid; i < 2048; i += 256) ((unsigned int*)h8)[i] = 0u;
  if (tid < 16) { y_buf[tid] = 0.0f; y_acc[tid] = 0.0f; }
  for (int n = tid; n < 1024; n += 256) {
    bias_l[n] = bih[n] + bhh[n];
    #pragma unroll
    for (int k = 0; k < 6; ++k) wih_l[n][k] = wih[n * 6 + k];
  }
  if (tid < 256) {
    const int f = tid;
    b1_l[f] = b1[f];
    w2_l[f] = w2[f];
    #pragma unroll
    for (int j = 0; j < 10; ++j) w1s_l[f][j] = w1[(size_t)f * 266 + 256 + j];
  }

  const float WS = 16.0f;
  v16i wB[4][2][2];

  #pragma unroll
  for (int gate = 0; gate < 4; ++gate) {
    #pragma unroll
    for (int half = 0; half < 2; ++half) {
      const int n = gate * 256 + w * 32 + half * 16 + ln;
      #pragma unroll
      for (int kc = 0; kc < 2; ++kc) {
        v16i r;
        #pragma unroll
        for (int v = 0; v < 16; ++v) {
          const int K = kc * 128 + (v >> 2) * 32 + g * 16 + (v & 3) * 4;
          const float* p = whh + (size_t)n * H_DIM + K;
          r[v] = (int)pk4_fp8(p[0] * WS, p[1] * WS, p[2] * WS, p[3] * WS);
        }
        wB[gate][half][kc] = r;
      }
    }
  }

  #pragma unroll
  for (int half = 0; half < 2; ++half) {
    const int f = w * 32 + half * 16 + ln;
    #pragma unroll
    for (int kc = 0; kc < 2; ++kc) {
      #pragma unroll
      for (int v = 0; v < 16; ++v) {
        const int K = kc * 128 + (v >> 2) * 32 + g * 16 + (v & 3) * 4;
        const float* p = w1 + (size_t)f * 266 + K;
        wFp[w][half][kc][l][v] = pk4_fp8(p[0] * WS, p[1] * WS, p[2] * WS, p[3] * WS);
      }
    }
  }
  const float b2s = b2[0];

  float creg[2][8];
  #pragma unroll
  for (int half = 0; half < 2; ++half)
    #pragma unroll
    for (int v = 0; v < 8; ++v) creg[half][v] = 0.0f;

  __syncthreads();

  for (int t = 0; t < S_LEN; ++t) {
    const int pbuf = t & 1;
    const int nbuf = pbuf ^ 1;

    if (tid < 16) {
      const float xv = x[(size_t)(b0 + tid) * S_LEN + t];
      const int   m  = mask[t];
      fullbuf[tid * 8 + 0] = (t == 0 || m != 0) ? xv : y_buf[tid];
    } else if (tid < 96) {
      const int q = tid - 16, r = q / 5, j = q - r * 5;
      fullbuf[r * 8 + 1 + j] = feat[((size_t)(b0 + r) * S_LEN + t) * 5 + j];
    } else {
      const int q = tid - 96, r = q / 10, j = q - r * 10;
      sbuf[r * 12 + j] = stat[((size_t)(b0 + r) * S_LEN + t) * 10 + j];
    }
    if (t + 1 < S_LEN && tid < 16) {
      __builtin_prefetch(&x[(size_t)(b0 + tid) * S_LEN + t + 1], 0, 3);
      __builtin_prefetch(&feat[((size_t)(b0 + tid) * S_LEN + t + 1) * 5], 0, 3);
      __builtin_prefetch(&stat[((size_t)(b0 + tid) * S_LEN + t + 1) * 10], 0, 3);
    }
    __syncthreads();

    v16i A0, A1;
    #pragma unroll
    for (int v = 0; v < 16; ++v) {
      const int K = (v >> 2) * 32 + g * 16 + (v & 3) * 4;
      A0[v] = (int)h8[pbuf][ln * 64 + (K >> 2)];
      A1[v] = (int)h8[pbuf][ln * 64 + 32 + (K >> 2)];
    }

    #pragma unroll
    for (int half = 0; half < 2; ++half) {
      v8f acc4[4];
      #pragma unroll
      for (int gate = 0; gate < 4; ++gate) {
        v8f a = {};
        a = __builtin_amdgcn_wmma_f32_16x16x128_fp8_fp8(A0, wB[gate][half][0], (short)0, a, false, false);
        a = __builtin_amdgcn_wmma_f32_16x16x128_fp8_fp8(A1, wB[gate][half][1], (short)0, a, false, false);
        acc4[gate] = a;
      }
      const int colbase = w * 32 + half * 16 + ln;
      float bia[4], wih_h[4][6];
      #pragma unroll
      for (int gate = 0; gate < 4; ++gate) {
        const int n = gate * 256 + colbase;
        bia[gate] = bias_l[n];
        #pragma unroll
        for (int k = 0; k < 6; ++k) wih_h[gate][k] = wih_l[n][k];
      }
      #pragma unroll
      for (int v = 0; v < 8; ++v) {
        const int row = g * 8 + v;
        float fl[6];
        #pragma unroll
        for (int k = 0; k < 6; ++k) fl[k] = fullbuf[row * 8 + k];
        float gv[4];
        #pragma unroll
        for (int gate = 0; gate < 4; ++gate) {
          float s = bia[gate];
          #pragma unroll
          for (int k = 0; k < 6; ++k) s += fl[k] * wih_h[gate][k];
          gv[gate] = s + 0.0625f * acc4[gate][v];
        }
        const float ig = sigm(gv[0]);
        const float fg = sigm(gv[1]);
        const float gg = tanh_fast(gv[2]);
        const float og = sigm(gv[3]);
        const float c  = fg * creg[half][v] + ig * gg;
        creg[half][v]  = c;
        const float h  = og * tanh_fast(c);
        ((unsigned char*)h8[nbuf])[row * 256 + colbase] = (unsigned char)fp8_e4m3(h);
      }
    }
    __syncthreads();

    v16i F0, F1;
    #pragma unroll
    for (int v = 0; v < 16; ++v) {
      const int K = (v >> 2) * 32 + g * 16 + (v & 3) * 4;
      F0[v] = (int)h8[nbuf][ln * 64 + (K >> 2)];
      F1[v] = (int)h8[nbuf][ln * 64 + 32 + (K >> 2)];
    }
    v8f fa[2];
    #pragma unroll
    for (int half = 0; half < 2; ++half) {
      v16i Bf0, Bf1;
      #pragma unroll
      for (int q = 0; q < 4; ++q) {
        const uint4 d0 = ((const uint4*)&wFp[w][half][0][l][0])[q];
        const uint4 d1 = ((const uint4*)&wFp[w][half][1][l][0])[q];
        Bf0[4*q+0] = (int)d0.x; Bf0[4*q+1] = (int)d0.y; Bf0[4*q+2] = (int)d0.z; Bf0[4*q+3] = (int)d0.w;
        Bf1[4*q+0] = (int)d1.x; Bf1[4*q+1] = (int)d1.y; Bf1[4*q+2] = (int)d1.z; Bf1[4*q+3] = (int)d1.w;
      }
      v8f a = {};
      a = __builtin_amdgcn_wmma_f32_16x16x128_fp8_fp8(F0, Bf0, (short)0, a, false, false);
      a = __builtin_amdgcn_wmma_f32_16x16x128_fp8_fp8(F1, Bf1, (short)0, a, false, false);
      fa[half] = a;
    }
    const int f0 = w * 32 + ln, f1 = f0 + 16;
    float w1s2[2][10], b12[2], w22[2];
    #pragma unroll
    for (int j = 0; j < 10; ++j) { w1s2[0][j] = w1s_l[f0][j]; w1s2[1][j] = w1s_l[f1][j]; }
    b12[0] = b1_l[f0]; b12[1] = b1_l[f1];
    w22[0] = w2_l[f0]; w22[1] = w2_l[f1];

    float part[8];
    #pragma unroll
    for (int v = 0; v < 8; ++v) {
      const int row = g * 8 + v;
      float sl[10];
      #pragma unroll
      for (int j = 0; j < 10; ++j) sl[j] = sbuf[row * 12 + j];
      float p = 0.0f;
      #pragma unroll
      for (int half = 0; half < 2; ++half) {
        float s = b12[half] + 0.0625f * fa[half][v];
        #pragma unroll
        for (int j = 0; j < 10; ++j) s += sl[j] * w1s2[half][j];
        s = fmaxf(s, 0.0f);
        p += s * w22[half];
      }
      part[v] = p;
    }
    #pragma unroll
    for (int v = 0; v < 8; ++v) {
      float p = part[v];
      p += __shfl_xor(p, 1, 16);
      p += __shfl_xor(p, 2, 16);
      p += __shfl_xor(p, 4, 16);
      p += __shfl_xor(p, 8, 16);
      if (ln == 0) atomicAdd(&y_acc[g * 8 + v], p);
    }
    __syncthreads();

    if (tid < 16) {
      const float yv = y_acc[tid] + b2s;
      out[(size_t)(b0 + tid) * S_LEN + t] = yv;
      y_buf[tid] = yv;
      y_acc[tid] = 0.0f;
    }
    __syncthreads();
  }
}

extern "C" void kernel_launch(void* const* d_in, const int* in_sizes, int n_in,
                              void* d_out, int out_size, void* d_ws, size_t ws_size,
                              hipStream_t stream) {
  (void)in_sizes; (void)n_in; (void)out_size; (void)d_ws; (void)ws_size;
  const float* x    = (const float*)d_in[0];
  const int*   mask = (const int*)  d_in[1];
  const float* feat = (const float*)d_in[2];
  const float* stat = (const float*)d_in[3];
  const float* wih  = (const float*)d_in[4];
  const float* whh  = (const float*)d_in[5];
  const float* bih  = (const float*)d_in[6];
  const float* bhh  = (const float*)d_in[7];
  const float* w1   = (const float*)d_in[8];
  const float* b1   = (const float*)d_in[9];
  const float* w2   = (const float*)d_in[10];
  const float* b2   = (const float*)d_in[11];
  float* out = (float*)d_out;

  dim3 grid(2048 / MTILE);
  dim3 block(256);
  lstm_persistent_fp8<<<grid, block, 0, stream>>>(x, mask, feat, stat, wih, whh,
                                                  bih, bhh, w1, b1, w2, b2, out);
}
